// TCFormerRegularAttention_28063316312342
// MI455X (gfx1250) — matrix operand order repaired and checked by reading, not compiled, not run on hardware
//
#include <hip/hip_runtime.h>

// ---------------------------------------------------------------------------
// TCFormer SR-attention, fused WMMA + TDM implementation for gfx1250 (MI455X).
// B=8, N=16384, C=64, heads=1, SR=8, N'=256, SCALE=1/8.
// ---------------------------------------------------------------------------

typedef __attribute__((ext_vector_type(16))) _Float16 v16h;
typedef __attribute__((ext_vector_type(8)))  _Float16 v8h;
typedef __attribute__((ext_vector_type(8)))  float    v8f;
typedef __attribute__((ext_vector_type(4)))  unsigned v4u;
typedef __attribute__((ext_vector_type(8)))  unsigned v8u;

#define BATCH 8
#define NTOK  16384
#define CH    64
#define NP    256          // reduced tokens (16x16)
#define SCALE 0.125f

// ---- WMMA wrapper ---------------------------------------------------------
__device__ __forceinline__ v8f wmma_f16(v16h a, v16h b, v8f c) {
  return __builtin_amdgcn_wmma_f32_16x16x32_f16(
      /*neg_a=*/false, a, /*neg_b=*/false, b,
      /*c_mod=*/(short)0, c, /*reuse_a=*/false, /*reuse_b=*/false);
}

// ---- Fragment loaders (per CDNA5 ISA 7.12.2 layouts) ----------------------
// A matrix 16x32 f16: lane L holds row M=L&15; halves 0..7 -> K = 8*(L>>4)+0..7,
// halves 8..15 -> K = 16 + 8*(L>>4)+0..7.  Source: row-major [M][K], stride in
// halves (multiple of 8 so 16B alignment holds).
__device__ __forceinline__ v16h load_A(const _Float16* base, int row0, int k0,
                                       int stride, int lane) {
  const _Float16* p = base + (size_t)(row0 + (lane & 15)) * stride + k0 + ((lane >> 4) << 3);
  v8h lo = *(const v8h*)(p);
  v8h hi = *(const v8h*)(p + 16);
  return __builtin_shufflevector(lo, hi, 0,1,2,3,4,5,6,7,8,9,10,11,12,13,14,15);
}

// B matrix 32x16 f16 (KxN): lane L holds column N=L&15; halves 0..7 -> K = 8*(L>>4)+0..7,
// halves 8..15 -> K = 16 + 8*(L>>4)+0..7.
// Source buffer is [N][K]-major (i.e. B transposed), stride in halves.
__device__ __forceinline__ v16h load_B(const _Float16* base, int k0, int n0,
                                       int stride, int lane) {
  const _Float16* p = base + (size_t)(n0 + (lane & 15)) * stride + k0 + ((lane >> 4) << 3);
  v8h lo = *(const v8h*)(p);
  v8h hi = *(const v8h*)(p + 16);
  return __builtin_shufflevector(lo, hi, 0,1,2,3,4,5,6,7,8,9,10,11,12,13,14,15);
}

// ---- TDM: 2D tile global f16 -> LDS with per-row padding ------------------
// rows x cols (halves); LDS row stride = cols + 8 halves (16B pad inserted by
// TDM pad engine).  pad_interval code: cols*2 bytes / 8 = 2^pi QWORDs.
__device__ __forceinline__ void tdm_load_2d(unsigned lds_off, const _Float16* gsrc,
                                            unsigned cols, unsigned rows,
                                            unsigned pad_interval_code) {
  unsigned long long ga = (unsigned long long)(uintptr_t)gsrc;
  v4u g0;
  g0[0] = 1u;                                   // count=1, user descriptor
  g0[1] = lds_off;                              // lds_addr
  g0[2] = (unsigned)ga;                         // global_addr[31:0]
  g0[3] = (unsigned)((ga >> 32) & 0x1FFFFFFu) | (2u << 30);  // addr[56:32] | type=2
  v8u g1;
  // pad_amount=3 (4 DWORDs = 16B), pad_enable, data_size=1 (2B), mask=0
  g1[0] = (3u << 25) | (pad_interval_code << 22) | (1u << 20) | (1u << 16);
  g1[1] = (cols & 0xFFFFu) << 16;               // tensor_dim0[15:0]
  g1[2] = (cols >> 16) | ((rows & 0xFFFFu) << 16);   // dim0[31:16] | dim1[15:0]
  g1[3] = (rows >> 16) | (cols << 16);          // dim1[31:16] | tile_dim0
  g1[4] = rows;                                 // tile_dim1 | tile_dim2=0
  g1[5] = cols;                                 // tensor_dim0_stride[31:0]
  g1[6] = 0u;                                   // stride0[47:32] | stride1[15:0]
  g1[7] = 0u;
  asm volatile("tensor_load_to_lds %0, %1" :: "s"(g0), "s"(g1) : "memory");
}

// ---------------------------------------------------------------------------
// Kernel 1: convert all weights fp32 -> fp16 into workspace.
// ---------------------------------------------------------------------------
__global__ void __launch_bounds__(256)
cvt_weights_kernel(const float* __restrict__ wq, const float* __restrict__ wkv,
                   const float* __restrict__ wsr, const float* __restrict__ wproj,
                   _Float16* __restrict__ wq_h, _Float16* __restrict__ wkv_h,
                   _Float16* __restrict__ wsr_h, _Float16* __restrict__ wproj_h) {
  int i = blockIdx.x * blockDim.x + threadIdx.x;
  if (i < CH * CH)     wq_h[i]    = (_Float16)wq[i];
  if (i < 2 * CH * CH) wkv_h[i]   = (_Float16)wkv[i];
  if (i < CH * CH)     wproj_h[i] = (_Float16)wproj[i];
  int total = CH * CH * 64;  // 64*64*8*8 = 262144
  for (int j = i; j < total; j += gridDim.x * blockDim.x)
    wsr_h[j] = (_Float16)wsr[j];
}

// ---------------------------------------------------------------------------
// Kernel 2: spatial-reduction conv (as K=4096 GEMM) + bias + LayerNorm + kv
// projection.  grid = B*16 blocks, 128 threads (4 waves).  Block handles 16
// reduced tokens; wave w owns conv output channels [16w,16w+16).
// Outputs: k_ws [B][256][64] f16 (row-major), vT_ws [B][64][256] f16.
// ---------------------------------------------------------------------------
__global__ void __launch_bounds__(128)
kv_reduce_kernel(const float* __restrict__ x, const _Float16* __restrict__ wsr_h,
                 const float* __restrict__ b_sr, const float* __restrict__ ln_g,
                 const float* __restrict__ ln_b, const _Float16* __restrict__ wkv_h,
                 _Float16* __restrict__ k_ws, _Float16* __restrict__ vT_ws) {
  __shared__ _Float16 aLDS[16 * 72];   // A chunk: 16 tokens x 64 spatial (padded)
  __shared__ float    lnLDS[16 * 64];  // conv output for LayerNorm
  __shared__ _Float16 a2LDS[16 * 72];  // LN'd tokens as A operand

  const int b    = blockIdx.x >> 4;
  const int tile = blockIdx.x & 15;    // 16 tokens per tile
  const int tid  = threadIdx.x;
  const int lane = tid & 31;
  const int wave = tid >> 5;
  const int n0   = wave << 4;          // conv c_out tile

  v8f acc = {};
  for (int cin = 0; cin < CH; ++cin) {
    // Stage A chunk: aLDS[t][s] = x[b][patch(tile*16+t, s)][cin]
#pragma unroll
    for (int j = 0; j < 8; ++j) {
      int idx = (tid << 3) + j;        // 0..1023
      int t = idx >> 6, s = idx & 63;
      int T = (tile << 4) + t;
      int ty = T >> 4, tx = T & 15;
      int kh = s >> 3, kw = s & 7;
      int n = ((ty << 3) + kh) * 128 + (tx << 3) + kw;
      aLDS[t * 72 + s] = (_Float16)x[((size_t)b * NTOK + n) * CH + cin];
    }
    __syncthreads();
    v16h a0 = load_A(aLDS, 0, 0, 72, lane);
    v16h a1 = load_A(aLDS, 0, 32, 72, lane);
    // w_sr_h is [c_out][c_in*64+s] row-major, stride 4096 -> natural [N][K]
    v16h b0 = load_B(wsr_h, cin * 64, n0, 4096, lane);
    v16h b1 = load_B(wsr_h, cin * 64 + 32, n0, 4096, lane);
    acc = wmma_f16(a0, b0, acc);
    acc = wmma_f16(a1, b1, acc);
    __syncthreads();
  }

  // bias, stash conv output f32 for LayerNorm
  {
    int c = n0 + (lane & 15);
    int mb = (lane >> 4) << 3;
    float bias = b_sr[c];
#pragma unroll
    for (int i = 0; i < 8; ++i) lnLDS[(mb + i) * 64 + c] = acc[i] + bias;
  }
  __syncthreads();

  // LayerNorm per token (threads 0..15, one token each)
  if (tid < 16) {
    float mu = 0.f;
    for (int c = 0; c < 64; ++c) mu += lnLDS[tid * 64 + c];
    mu *= (1.f / 64.f);
    float var = 0.f;
    for (int c = 0; c < 64; ++c) { float d = lnLDS[tid * 64 + c] - mu; var += d * d; }
    var *= (1.f / 64.f);
    float rs = rsqrtf(var + 1e-5f);
    for (int c = 0; c < 64; ++c)
      a2LDS[tid * 72 + c] = (_Float16)((lnLDS[tid * 64 + c] - mu) * rs * ln_g[c] + ln_b[c]);
  }
  __syncthreads();

  // kv projection: 128 outputs = 8 N-tiles; wave handles tiles {2w, 2w+1}
  v16h pa0 = load_A(a2LDS, 0, 0, 72, lane);
  v16h pa1 = load_A(a2LDS, 0, 32, 72, lane);
#pragma unroll
  for (int t2 = 0; t2 < 2; ++t2) {
    int np = (wave << 5) + (t2 << 4);
    v8f accp = {};
    v16h bb0 = load_B(wkv_h, 0, np, 64, lane);
    v16h bb1 = load_B(wkv_h, 32, np, 64, lane);
    accp = wmma_f16(pa0, bb0, accp);
    accp = wmma_f16(pa1, bb1, accp);
    int j = np + (lane & 15);
    int mb = (lane >> 4) << 3;
    if (j < 64) {  // k: row-major [token][c]
#pragma unroll
      for (int i = 0; i < 8; ++i)
        k_ws[((size_t)b * NP + (tile << 4) + mb + i) * CH + j] = (_Float16)accp[i];
    } else {       // v: transposed [c][token]
      int jv = j - 64;
#pragma unroll
      for (int i = 0; i < 8; ++i)
        vT_ws[((size_t)b * CH + jv) * NP + (tile << 4) + mb + i] = (_Float16)accp[i];
    }
  }
}

// ---------------------------------------------------------------------------
// Kernel 3: fused q-proj -> S=q k^T -> softmax -> O=P v -> out-proj.
// grid = B * (16384/64) = 2048 blocks, 128 threads (4 waves).
// Each block: 64 query tokens, full 256-wide attention (k,vT in LDS).
// Wave w owns query rows [16w, 16w+16).  k/vT/w_q/w_proj staged by TDM.
// ---------------------------------------------------------------------------
// LDS layout in halves:
#define OFF_WQ 0
#define OFF_WP (OFF_WQ + 64 * 72)
#define OFF_K  (OFF_WP + 64 * 72)
#define OFF_VT (OFF_K + 256 * 72)
#define OFF_X  (OFF_VT + 64 * 264)
#define OFF_Q  (OFF_X + 64 * 72)
#define OFF_S  (OFF_Q + 64 * 72)
#define OFF_O  (OFF_S + 64 * 264)
#define OFF_END (OFF_O + 64 * 72)
#define ATTN_SMEM_BYTES (OFF_END * 2 + 320 * 4)

__global__ void __launch_bounds__(128)
attn_kernel(const float* __restrict__ x, const _Float16* __restrict__ wq_h,
            const _Float16* __restrict__ wproj_h, const _Float16* __restrict__ k_ws,
            const _Float16* __restrict__ vT_ws, const float* __restrict__ b_proj,
            float* __restrict__ out) {
  extern __shared__ _Float16 smem[];
  _Float16* sWq = smem + OFF_WQ;   // [64][72]
  _Float16* sWp = smem + OFF_WP;   // [64][72]
  _Float16* sK  = smem + OFF_K;    // [256][72]
  _Float16* sVT = smem + OFF_VT;   // [64][264]
  _Float16* sX  = smem + OFF_X;    // [64][72]
  _Float16* sQ  = smem + OFF_Q;    // [64][72]
  _Float16* sS  = smem + OFF_S;    // [64][264]
  _Float16* sO  = smem + OFF_O;    // [64][72]
  float* fsc = (float*)(smem + OFF_END);  // [0,128): pmax  [128,256): psum  [256,320): rowsum

  const int b    = blockIdx.x >> 8;          // batch
  const int m0   = (blockIdx.x & 255) << 6;  // query tile base
  const int tid  = threadIdx.x;
  const int lane = tid & 31;
  const int wave = tid >> 5;
  const int mrow = wave << 4;
  const int mb   = (lane >> 4) << 3;
  const int ncol = lane & 15;

  // ---- async tile staging via Tensor Data Mover (one D# per wave) ----
  // k : 256 rows x 64 halves,  pad 32DW->+4DW  (stride 72)   pi=4
  // vT: 64 rows x 256 halves,  pad 128DW->+4DW (stride 264)  pi=6
  // wq/wp: 64 rows x 64 halves                               pi=4
  if (wave == 0) {
    tdm_load_2d((unsigned)(uintptr_t)sK, k_ws + (size_t)b * NP * CH, 64, 256, 4u);
  } else if (wave == 1) {
    tdm_load_2d((unsigned)(uintptr_t)sVT, vT_ws + (size_t)b * CH * NP, 256, 64, 6u);
  } else if (wave == 2) {
    tdm_load_2d((unsigned)(uintptr_t)sWq, wq_h, 64, 64, 4u);
  } else {
    tdm_load_2d((unsigned)(uintptr_t)sWp, wproj_h, 64, 64, 4u);
  }

  // ---- x tile f32 -> f16 (VALU work, overlaps the TDM DMA) ----
  {
    const float4* gx = (const float4*)(x + ((size_t)b * NTOK + m0) * CH);
    for (int i = tid; i < 1024; i += 128) {           // 64x64 in 4-float chunks
      int r = i >> 4, c = (i & 15) << 2;
      float4 v = gx[i];
      _Float16* d = sX + r * 72 + c;
      d[0] = (_Float16)v.x; d[1] = (_Float16)v.y;
      d[2] = (_Float16)v.z; d[3] = (_Float16)v.w;
    }
  }
  __builtin_amdgcn_s_wait_tensorcnt(0);
  __syncthreads();

  // ---- q = (x @ w_q^T) * SCALE ----
  {
    v16h xa0 = load_A(sX, mrow, 0, 72, lane);
    v16h xa1 = load_A(sX, mrow, 32, 72, lane);
#pragma unroll
    for (int nt = 0; nt < 4; ++nt) {
      v8f acc = {};
      acc = wmma_f16(xa0, load_B(sWq, 0, nt << 4, 72, lane), acc);
      acc = wmma_f16(xa1, load_B(sWq, 32, nt << 4, 72, lane), acc);
#pragma unroll
      for (int i = 0; i < 8; ++i)
        sQ[(mrow + mb + i) * 72 + (nt << 4) + ncol] = (_Float16)(acc[i] * SCALE);
    }
  }
  __syncthreads();

  // ---- S = q @ k^T  (64 x 256) ----
  {
    v16h qa0 = load_A(sQ, mrow, 0, 72, lane);
    v16h qa1 = load_A(sQ, mrow, 32, 72, lane);
#pragma unroll
    for (int nt = 0; nt < 16; ++nt) {
      v8f acc = {};
      acc = wmma_f16(qa0, load_B(sK, 0, nt << 4, 72, lane), acc);
      acc = wmma_f16(qa1, load_B(sK, 32, nt << 4, 72, lane), acc);
#pragma unroll
      for (int i = 0; i < 8; ++i)
        sS[(mrow + mb + i) * 264 + (nt << 4) + ncol] = (_Float16)acc[i];
    }
  }
  __syncthreads();

  // ---- exact softmax over 256 cols; P = exp(S - max) left unnormalized,
  //      1/rowsum folded into the O epilogue ----
  {
    int r = tid & 63, hh = tid >> 6;
    const _Float16* row = sS + r * 264 + (hh << 7);
    float m = -1e30f;
    for (int c = 0; c < 128; ++c) m = fmaxf(m, (float)row[c]);
    fsc[(hh << 6) + r] = m;
    __syncthreads();
    float rowm = fmaxf(fsc[r], fsc[64 + r]);
    _Float16* roww = sS + r * 264 + (hh << 7);
    float s = 0.f;
    for (int c = 0; c < 128; ++c) {
      float e = __expf((float)roww[c] - rowm);
      s += e;
      roww[c] = (_Float16)e;
    }
    fsc[128 + (hh << 6) + r] = s;
    __syncthreads();
    if (tid < 64) fsc[256 + tid] = fsc[128 + tid] + fsc[192 + tid];
    __syncthreads();
  }

  // ---- O = P @ v  (K = 256), divide by rowsum ----
  {
    v16h aS[8];
#pragma unroll
    for (int kt = 0; kt < 8; ++kt) aS[kt] = load_A(sS, mrow, kt << 5, 264, lane);
#pragma unroll
    for (int dt = 0; dt < 4; ++dt) {
      v8f acc = {};
#pragma unroll
      for (int kt = 0; kt < 8; ++kt)
        acc = wmma_f16(aS[kt], load_B(sVT, kt << 5, dt << 4, 264, lane), acc);
#pragma unroll
      for (int i = 0; i < 8; ++i) {
        int m = mrow + mb + i;
        float inv = 1.f / fsc[256 + m];
        sO[m * 72 + (dt << 4) + ncol] = (_Float16)(acc[i] * inv);
      }
    }
  }
  __syncthreads();

  // ---- out = O @ w_proj^T + b_proj ----
  {
    v16h oa0 = load_A(sO, mrow, 0, 72, lane);
    v16h oa1 = load_A(sO, mrow, 32, 72, lane);
#pragma unroll
    for (int nt = 0; nt < 4; ++nt) {
      v8f acc = {};
      acc = wmma_f16(oa0, load_B(sWp, 0, nt << 4, 72, lane), acc);
      acc = wmma_f16(oa1, load_B(sWp, 32, nt << 4, 72, lane), acc);
      int cc = (nt << 4) + ncol;
      float bias = b_proj[cc];
#pragma unroll
      for (int i = 0; i < 8; ++i)
        out[((size_t)b * NTOK + m0 + mrow + mb + i) * CH + cc] = acc[i] + bias;
    }
  }
}

// ---------------------------------------------------------------------------
extern "C" void kernel_launch(void* const* d_in, const int* in_sizes, int n_in,
                              void* d_out, int out_size, void* d_ws, size_t ws_size,
                              hipStream_t stream) {
  const float* x      = (const float*)d_in[0];
  // d_in[1]=H, d_in[2]=W (fixed 128, unused)
  const float* w_q    = (const float*)d_in[3];
  const float* w_kv   = (const float*)d_in[4];
  const float* w_sr   = (const float*)d_in[5];
  const float* b_sr   = (const float*)d_in[6];
  const float* ln_g   = (const float*)d_in[7];
  const float* ln_b   = (const float*)d_in[8];
  const float* w_proj = (const float*)d_in[9];
  const float* b_proj = (const float*)d_in[10];
  float* out = (float*)d_out;

  char* ws = (char*)d_ws;
  _Float16* wq_h    = (_Float16*)ws; ws += (size_t)CH * CH * 2;
  _Float16* wproj_h = (_Float16*)ws; ws += (size_t)CH * CH * 2;
  _Float16* wkv_h   = (_Float16*)ws; ws += (size_t)2 * CH * CH * 2;
  _Float16* wsr_h   = (_Float16*)ws; ws += (size_t)CH * CH * 64 * 2;
  _Float16* k_ws    = (_Float16*)ws; ws += (size_t)BATCH * NP * CH * 2;
  _Float16* vT_ws   = (_Float16*)ws; ws += (size_t)BATCH * CH * NP * 2;

  cvt_weights_kernel<<<1024, 256, 0, stream>>>(w_q, w_kv, w_sr, w_proj,
                                               wq_h, wkv_h, wsr_h, wproj_h);

  kv_reduce_kernel<<<BATCH * 16, 128, 0, stream>>>(x, wsr_h, b_sr, ln_g, ln_b,
                                                   wkv_h, k_ws, vT_ws);

  attn_kernel<<<BATCH * (NTOK / 64), 128, ATTN_SMEM_BYTES, stream>>>(
      x, wq_h, wproj_h, k_ws, vT_ws, b_proj, out);
}
